// HeteroGNNGraphPredictor_89644557402629
// MI455X (gfx1250) — matrix operand order repaired and checked by reading, not compiled, not run on hardware
//
#include <hip/hip_runtime.h>
#include <hip/hip_bf16.h>

// ---------------------------------------------------------------------------
// HeteroGNN: 2x hetero GraphConv (norm='both') + mean_nodes + MLP head.
//   * degrees once; rsqrt(clamp(deg,1)) in place
//   * edge pass: one wave32 per edge, float4 per lane, f32 atomics (HBM/L2 bound)
//   * weights repacked once -> bf16 in WMMA B-fragment order (contiguous b128)
//   * GEMM: A tile scaled+bf16-converted into LDS once per block (fragment
//     order), 8 waves consume it; V_WMMA_F32_16X16X32_BF16, K=128 -> 4 WMMA
//     per relation per 16x16 tile; bias+ReLU fused in epilogue
//   * layer-2 'tr' relation skipped (dead in reference output)
// ---------------------------------------------------------------------------

#define N_TILE 50000
#define N_RR   50000
#define D      128
#define E_TT   800000
#define E_RT   400000
#define E_TR   400000
#define N_G    16

typedef __attribute__((ext_vector_type(16))) __bf16 v16bf;
typedef __attribute__((ext_vector_type(8)))  __bf16 v8bf;
typedef __attribute__((ext_vector_type(8)))  float  v8f;

// ---------------------------------- utility kernels -------------------------

__global__ void zero_kernel(float* __restrict__ p, long long n) {
  long long i = (long long)blockIdx.x * blockDim.x + threadIdx.x;
  if (i < n) p[i] = 0.0f;
}

__global__ void edge_deg_kernel(const int* __restrict__ src, const int* __restrict__ dst,
                                float* __restrict__ outdeg, float* __restrict__ indeg,
                                int nedges) {
  int e = blockIdx.x * blockDim.x + threadIdx.x;
  if (e >= nedges) return;
  atomicAdd(&outdeg[src[e]], 1.0f);
  atomicAdd(&indeg[dst[e]], 1.0f);
}

__global__ void rsq_kernel(float* __restrict__ v, int n) {
  int i = blockIdx.x * blockDim.x + threadIdx.x;
  if (i < n) v[i] = rsqrtf(fmaxf(v[i], 1.0f));
}

// agg[dst] += x[src] * rsq_out[src]; one wave per edge, one float4 per lane.
__global__ void edge_scatter_kernel(const int* __restrict__ src, const int* __restrict__ dst,
                                    const float* __restrict__ x, const float* __restrict__ rsq_out,
                                    float* __restrict__ agg, int nedges) {
  long long tid = (long long)blockIdx.x * blockDim.x + threadIdx.x;
  int e    = (int)(tid >> 5);
  int lane = (int)(tid & 31);
  if (e >= nedges) return;
  int s = src[e];
  int t = dst[e];
  float r = rsq_out[s];
  const float4 xv = *(const float4*)&x[(size_t)s * D + lane * 4];
  float* dp = &agg[(size_t)t * D + lane * 4];
  atomicAdd(dp + 0, xv.x * r);
  atomicAdd(dp + 1, xv.y * r);
  atomicAdd(dp + 2, xv.z * r);
  atomicAdd(dp + 3, xv.w * r);
}

// Repack a 128x128 row-major f32 weight matrix into bf16 laid out in this kernel's own
// B-fragment order (contiguous k, matched by the A tile): frag index = ((kb4*8 + wave)*32 + lane), 16 elems
// each; element 2v+p of a fragment is W[kb4*32 + 16*(lane>>4) + 2v + p][wave*16 + (lane&15)].
__global__ void repack_w_kernel(const float* __restrict__ W, __bf16* __restrict__ Wp) {
  int g = blockIdx.x * blockDim.x + threadIdx.x;   // 0 .. 16383
  if (g >= D * D) return;
  int pos  = g & 15;         // 2v + p
  int frag = g >> 4;
  int lane = frag & 31;
  int wv   = (frag >> 5) & 7;
  int kb4  = frag >> 8;      // 0..3
  int k   = kb4 * 32 + 16 * (lane >> 4) + pos;
  int col = wv * 16 + (lane & 15);
  Wp[g] = (__bf16)W[(size_t)k * D + col];
}

// -------------------------- fused WMMA GEMM kernel ---------------------------
// out[row,col] = relu( sum_rel((rsq_rel[row]*A_rel[row,:]) @ W_rel)[col] + bias[col] )
// Block = 256 threads (8 wave32), handles 16 rows x 128 cols; wave w owns
// column tile w. A tile staged in LDS (scaled, bf16, fragment order) once.
// nrows multiple of 16 (50000 = 3125*16) -> EXEC all-ones for WMMA.
__global__ __launch_bounds__(256)
void gemm_wmma_kernel(const float* __restrict__ A1, const float* __restrict__ rsq1,
                      const __bf16* __restrict__ Wp1, const float* __restrict__ b1,
                      const float* __restrict__ A2, const float* __restrict__ rsq2,
                      const __bf16* __restrict__ Wp2, const float* __restrict__ b2,
                      float* __restrict__ out) {
  __shared__ __bf16 As[2][4 * 32 * 16];        // [rel][frag(kb4,lane)][16], 4KB each

  const int tid  = threadIdx.x;
  const int lane = tid & 31;
  const int wave = tid >> 5;                   // column tile 0..7
  const int row0 = blockIdx.x * 16;
  const int col0 = wave * 16;
  const int m    = lane & 15;
  const int half = lane >> 4;

  // ---- stage A fragments: 256 threads cover 128 frags x 2 halves ----
  {
    const int frag = tid >> 1;                 // 0..127
    const int part = tid & 1;                  // low/high 8 elems of fragment
    const int fl   = frag & 31;                // fragment's lane
    const int fkb  = frag >> 5;                // 0..3 (K block of 32)
    const int fm   = fl & 15;
    const int fh   = fl >> 4;
    #pragma unroll
    for (int rel = 0; rel < 2; ++rel) {
      const float* A = rel ? A2 : A1;
      const float* R = rel ? rsq2 : rsq1;
      if (A == nullptr) break;                 // uniform branch
      const float scale = R[row0 + fm];
      const float* arow = A + (size_t)(row0 + fm) * D;
      v8bf t;
      #pragma unroll
      for (int i = 0; i < 4; ++i) {
        int v = part * 4 + i;                  // VGPR slot 0..7
        int k = fkb * 32 + 16 * fh + 2 * v;
        t[2 * i]     = (__bf16)(arow[k]     * scale);
        t[2 * i + 1] = (__bf16)(arow[k + 1] * scale);
      }
      *(v8bf*)&As[rel][frag * 16 + part * 8] = t;   // 16B ds_store
    }
  }
  __syncthreads();

  // ---- WMMA main loop ----
  v8f acc = {};
  #pragma unroll
  for (int rel = 0; rel < 2; ++rel) {
    const __bf16* Wp = rel ? Wp2 : Wp1;
    if ((rel ? A2 : A1) == nullptr) break;     // uniform branch
    #pragma unroll
    for (int kb4 = 0; kb4 < 4; ++kb4) {
      v16bf a = *(const v16bf*)&As[rel][(kb4 * 32 + lane) * 16];                 // 2x ds_load_b128
      v16bf b = *(const v16bf*)&Wp[((size_t)(kb4 * 8 + wave) * 32 + lane) * 16]; // 2x global_load_b128
      acc = __builtin_amdgcn_wmma_f32_16x16x32_bf16(
          /*neg_a=*/false, a, /*neg_b=*/false, b,
          /*c_mod=*/(short)0, acc, /*reuse_a=*/false, /*reuse_b=*/false);
    }
  }

  // ---- epilogue: bias + relu, C/D layout: VGPR r -> row M=r (lanes 0-15) / M=8+r ----
  const float bias = (b1 ? b1[col0 + m] : 0.0f) + (b2 ? b2[col0 + m] : 0.0f);
  #pragma unroll
  for (int r = 0; r < 8; ++r) {
    int row = row0 + half * 8 + r;
    out[(size_t)row * D + (col0 + m)] = fmaxf(acc[r] + bias, 0.0f);
  }
}

// ---------------------------- pooling + MLP head -----------------------------

__global__ void pool_kernel(const float* __restrict__ h, const int* __restrict__ gid,
                            float* __restrict__ pool, float* __restrict__ cnt, int n) {
  long long tid = (long long)blockIdx.x * blockDim.x + threadIdx.x;
  int node = (int)(tid >> 5);
  int lane = (int)(tid & 31);
  if (node >= n) return;
  int g = gid[node];
  const float4 hv = *(const float4*)&h[(size_t)node * D + lane * 4];
  float* dp = &pool[(size_t)g * D + lane * 4];
  atomicAdd(dp + 0, hv.x);
  atomicAdd(dp + 1, hv.y);
  atomicAdd(dp + 2, hv.z);
  atomicAdd(dp + 3, hv.w);
  if (lane == 0) atomicAdd(&cnt[g], 1.0f);
}

__global__ __launch_bounds__(128)
void mlp_kernel(const float* __restrict__ pool, const float* __restrict__ cnt,
                const float* __restrict__ Wm1, const float* __restrict__ bm1,
                const float* __restrict__ Wm2, const float* __restrict__ bm2,
                float* __restrict__ out) {
  __shared__ float hg[N_G][D];
  __shared__ float hid[N_G][D];
  int j = threadIdx.x;
  for (int g = 0; g < N_G; ++g)
    hg[g][j] = pool[(size_t)g * D + j] / fmaxf(cnt[g], 1.0f);
  __syncthreads();
  for (int g = 0; g < N_G; ++g) {
    float acc = bm1[j];
    for (int d = 0; d < D; ++d) acc += hg[g][d] * Wm1[(size_t)d * D + j];
    hid[g][j] = fmaxf(acc, 0.0f);
  }
  __syncthreads();
  if (j < N_G) {
    float acc = bm2[0];
    for (int k = 0; k < D; ++k) acc += hid[j][k] * Wm2[k];
    out[j] = acc;
  }
}

// --------------------------------- launcher ---------------------------------

extern "C" void kernel_launch(void* const* d_in, const int* in_sizes, int n_in,
                              void* d_out, int out_size, void* d_ws, size_t ws_size,
                              hipStream_t stream) {
  const float* x_tile  = (const float*)d_in[0];
  const float* x_rr    = (const float*)d_in[1];
  const int*   tt_src  = (const int*)d_in[2];
  const int*   tt_dst  = (const int*)d_in[3];
  const int*   rt_src  = (const int*)d_in[4];
  const int*   rt_dst  = (const int*)d_in[5];
  const int*   tr_src  = (const int*)d_in[6];
  const int*   tr_dst  = (const int*)d_in[7];
  const int*   tile_gid= (const int*)d_in[8];
  const float* W1_tt = (const float*)d_in[9],  *b1_tt = (const float*)d_in[10];
  const float* W1_rt = (const float*)d_in[11], *b1_rt = (const float*)d_in[12];
  const float* W1_tr = (const float*)d_in[13], *b1_tr = (const float*)d_in[14];
  const float* W2_tt = (const float*)d_in[15], *b2_tt = (const float*)d_in[16];
  const float* W2_rt = (const float*)d_in[17], *b2_rt = (const float*)d_in[18];
  // d_in[19..20] = W2_tr/b2_tr: dead in reference output, skipped.
  const float* Wm1 = (const float*)d_in[21], *bm1 = (const float*)d_in[22];
  const float* Wm2 = (const float*)d_in[23], *bm2 = (const float*)d_in[24];
  float* out = (float*)d_out;

  // ---- workspace layout ----
  float* ws = (float*)d_ws;
  const long long NF = (long long)N_TILE * D;       // 6.4M floats per feature buffer
  float* rsq_out_tt = ws + 0;                       // N_TILE
  float* rsq_in_tt  = ws + 50000;                   // N_TILE
  float* rsq_out_rt = ws + 100000;                  // N_RR
  float* rsq_in_rt  = ws + 150000;                  // N_TILE
  float* rsq_out_tr = ws + 200000;                  // N_TILE
  float* rsq_in_tr  = ws + 250000;                  // N_RR
  float* agg_a  = ws + 300000;                      // [N_TILE, D]
  float* agg_b  = agg_a + NF;                       // [N_TILE, D]
  float* agg_c  = agg_b + NF;                       // [N_RR, D]
  float* h_tile = agg_c + NF;                       // [N_TILE, D]
  float* h_rr   = h_tile + NF;                      // [N_RR, D]
  float* h2_tile= h_rr + NF;                        // [N_TILE, D]
  float* pool   = h2_tile + NF;                     // [N_G, D]
  float* cnt    = pool + N_G * D;                   // [N_G]
  __bf16* wp0 = (__bf16*)(cnt + N_G);               // 5 x 16384 bf16 repacked weights
  __bf16* wp1 = wp0 + D * D;
  __bf16* wp2 = wp1 + D * D;
  __bf16* wp3 = wp2 + D * D;
  __bf16* wp4 = wp3 + D * D;

  auto cdiv = [](long long a, long long b) { return (int)((a + b - 1) / b); };
  const int T = 256;

  // 1) zero degrees + layer-1 agg buffers (contiguous region)
  long long z1 = 300000 + 3 * NF;
  zero_kernel<<<cdiv(z1, T), T, 0, stream>>>(ws, z1);

  // 2) repack the 5 live weight matrices to bf16 fragment order
  repack_w_kernel<<<cdiv(D * D, T), T, 0, stream>>>(W1_tt, wp0);
  repack_w_kernel<<<cdiv(D * D, T), T, 0, stream>>>(W1_rt, wp1);
  repack_w_kernel<<<cdiv(D * D, T), T, 0, stream>>>(W1_tr, wp2);
  repack_w_kernel<<<cdiv(D * D, T), T, 0, stream>>>(W2_tt, wp3);
  repack_w_kernel<<<cdiv(D * D, T), T, 0, stream>>>(W2_rt, wp4);

  // 3) degrees (shared by both layers) + rsqrt(clamp(.,1))
  edge_deg_kernel<<<cdiv(E_TT, T), T, 0, stream>>>(tt_src, tt_dst, rsq_out_tt, rsq_in_tt, E_TT);
  edge_deg_kernel<<<cdiv(E_RT, T), T, 0, stream>>>(rt_src, rt_dst, rsq_out_rt, rsq_in_rt, E_RT);
  edge_deg_kernel<<<cdiv(E_TR, T), T, 0, stream>>>(tr_src, tr_dst, rsq_out_tr, rsq_in_tr, E_TR);
  rsq_kernel<<<cdiv(300000, T), T, 0, stream>>>(ws, 300000);

  // 4) layer-1 edge aggregation (one wave per edge)
  edge_scatter_kernel<<<cdiv((long long)E_TT * 32, T), T, 0, stream>>>(tt_src, tt_dst, x_tile, rsq_out_tt, agg_a, E_TT);
  edge_scatter_kernel<<<cdiv((long long)E_RT * 32, T), T, 0, stream>>>(rt_src, rt_dst, x_rr,   rsq_out_rt, agg_b, E_RT);
  edge_scatter_kernel<<<cdiv((long long)E_TR * 32, T), T, 0, stream>>>(tr_src, tr_dst, x_tile, rsq_out_tr, agg_c, E_TR);

  // 5) layer-1 dense: fused (tt + rt) -> h_tile, tr -> h_rr
  gemm_wmma_kernel<<<N_TILE / 16, 256, 0, stream>>>(
      agg_a, rsq_in_tt, wp0, b1_tt, agg_b, rsq_in_rt, wp1, b1_rt, h_tile);
  gemm_wmma_kernel<<<N_RR / 16, 256, 0, stream>>>(
      agg_c, rsq_in_tr, wp2, b1_tr, nullptr, nullptr, nullptr, nullptr, h_rr);

  // 6) layer-2: re-zero agg_a/agg_b, scatter, fused GEMM -> h2_tile
  zero_kernel<<<cdiv(2 * NF, T), T, 0, stream>>>(agg_a, 2 * NF);
  edge_scatter_kernel<<<cdiv((long long)E_TT * 32, T), T, 0, stream>>>(tt_src, tt_dst, h_tile, rsq_out_tt, agg_a, E_TT);
  edge_scatter_kernel<<<cdiv((long long)E_RT * 32, T), T, 0, stream>>>(rt_src, rt_dst, h_rr,   rsq_out_rt, agg_b, E_RT);
  gemm_wmma_kernel<<<N_TILE / 16, 256, 0, stream>>>(
      agg_a, rsq_in_tt, wp3, b2_tt, agg_b, rsq_in_rt, wp4, b2_rt, h2_tile);

  // 7) mean-pool per graph + MLP head
  zero_kernel<<<cdiv(N_G * D + N_G, T), T, 0, stream>>>(pool, N_G * D + N_G);
  pool_kernel<<<cdiv((long long)N_TILE * 32, T), T, 0, stream>>>(h2_tile, tile_gid, pool, cnt, N_TILE);
  mlp_kernel<<<1, 128, 0, stream>>>(pool, cnt, Wm1, bm1, Wm2, bm2, out);
}
